// FilterBank_39135742001705
// MI455X (gfx1250) — matrix operand order repaired and checked by reading, not compiled, not run on hardware
//
#include <hip/hip_runtime.h>
#include <hip/hip_bf16.h>

// FIR band-pass (413 taps, zero-phase, reflect_limited padding) as WMMA GEMM.
//
// out[row][t] = sum_k h[k] * xe_row(t + 206 - k)
// GEMM tile: D(16x16) += A(16x32) x B(32x16), A = Toeplitz(h) in f16 (constant
// across all tiles, kept in VGPRs), B = f16 extended signal window, C/D = f32.
//
// d_ws layout:
//   [0,       14336)   : packed f16 Toeplitz A tiles (14 chunks x 32 lanes x 16 halves)
//   [16384, 16384+2048*30848*2) : f16 reflect-extended signal, row stride 30848 halves
// (~126.4 MB total; resident in the 192 MB L2 so the overlapped window reads
//  of the main kernel stream from L2, not HBM.)

typedef __attribute__((ext_vector_type(16))) _Float16 v16h;
typedef __attribute__((ext_vector_type(2)))  _Float16 v2h;
typedef __attribute__((ext_vector_type(8)))  float    v8f;

#define NT    30000      // samples per row
#define NROWS 2048       // 32*64
#define NTAPS 413
#define NE    412        // edge pad width = NTAPS-1
#define NCHUNK 14        // ceil((NE+16)/32) = 448/32 K-chunks
#define XSTRIDE 30848    // padded row stride in halves (multiple of 64)
#define XOFF  2          // phase shift so B loads are 32B-aligned (206+2=208=13*16)
#define TTILES 1875      // 30000/16 time tiles per row
#define WSLOTS 120       // wave slots per row tile (gridDim.y=15 * 8 waves)

// ---- prep 1: pack f16 Toeplitz filter tiles in exact WMMA A-layout ---------
// A[i,k] = h[i + 412 - k], chunk c covers k in [32c, 32c+32).
// A lane layout used here (contiguous k, same as B loads): lanes 0-15 hold M=i with half e ->
// K = e ; lanes 16-31 hold same M rows with K offset +16.
__global__ void FilterBank_prep_h(const float* __restrict__ h,
                                  _Float16* __restrict__ hT) {
    int id = blockIdx.x * 256 + threadIdx.x;      // 0 .. 14*512-1
    if (id >= NCHUNK * 512) return;
    int c = id >> 9;
    int l = (id >> 4) & 31;
    int e = id & 15;
    int kk = e + ((l & 16) ? 16 : 0);             // K within chunk, 0..31
    int i  = l & 15;                              // M row (time offset)
    int idx = i + NE - (32 * c + kk);
    float v = (idx >= 0 && idx < NTAPS) ? h[idx] : 0.0f;
    hT[id] = (_Float16)v;
}

// ---- prep 2: reflect_limited-extended f16 signal ---------------------------
// buffer[row][pos] = f16( xe(pos - XOFF - NE) ), zero outside support.
__global__ void FilterBank_prep_xe(const float* __restrict__ x,
                                   _Float16* __restrict__ xe) {
    int row = blockIdx.y;
    int w = blockIdx.x * blockDim.x + threadIdx.x;   // 32-bit word index
    if (w >= XSTRIDE / 2) return;
    const float* xr = x + (size_t)row * NT;
    v2h out2;
    #pragma unroll
    for (int s = 0; s < 2; ++s) {
        int pos = 2 * w + s;
        int t = pos - XOFF - NE;                     // signal-domain index
        float val;
        if (t < 0) {
            int m = -t;
            val = (m <= NE) ? (2.0f * xr[0] - xr[m]) : 0.0f;
        } else if (t < NT) {
            val = xr[t];
        } else {
            int m = t - (NT - 1);
            val = (m <= NE) ? (2.0f * xr[NT - 1] - xr[NT - 1 - m]) : 0.0f;
        }
        out2[s] = (_Float16)val;
    }
    *(v2h*)(xe + (size_t)row * XSTRIDE + 2 * w) = out2;
}

// ---- main: WMMA FIR --------------------------------------------------------
__global__ void __launch_bounds__(256)
FilterBank_fir_wmma(const _Float16* __restrict__ xe,
                    const _Float16* __restrict__ hT,
                    float* __restrict__ out) {
    const int lane = threadIdx.x & 31;
    const int wave = threadIdx.x >> 5;                 // 0..7
    const int rowTile = blockIdx.x;                    // 0..127
    const int waveSlot = blockIdx.y * 8 + wave;        // 0..119

    // Filter Toeplitz tiles: loaded once, live in VGPRs for all time tiles.
    v16h A[NCHUNK];
    #pragma unroll
    for (int c = 0; c < NCHUNK; ++c)
        A[c] = *(const v16h*)(hT + c * 512 + lane * 16);

    const int ch  = rowTile * 16 + (lane & 15);        // channel row (N)
    const int lhi = (lane >> 4) & 1;                   // 0: K/M low half, 1: high
    const _Float16* xrow = xe + (size_t)ch * XSTRIDE;

    // Strength-reduced pointers: advance by WSLOTS*16 elements per iteration.
    const _Float16* bptr = xrow + waveSlot * 16 + 206 + XOFF + lhi * 16; // 32B-aligned
    float* optr = out + (size_t)ch * NT + lhi * 8 + waveSlot * 16;

    for (int tt = waveSlot; tt < TTILES; tt += WSLOTS) {
        // Preload ALL 14 B chunks into distinct registers. The sched_barrier
        // pins every global_load_b128 ahead of the WMMA chain, so the chain
        // drains with graduated s_wait_loadcnt: one L2 latency per tile
        // instead of one per chunk. (~240 VGPRs live: A=112, B=112, acc=8.)
        v16h B[NCHUNK];
        #pragma unroll
        for (int c = 0; c < NCHUNK; ++c)
            B[c] = *(const v16h*)(bptr + 32 * c);

#if defined(__has_builtin)
#if __has_builtin(__builtin_amdgcn_sched_barrier)
        __builtin_amdgcn_sched_barrier(0);   // nothing crosses: loads stay above
#endif
#endif

        v8f acc = {};
        #pragma unroll
        for (int c = 0; c < NCHUNK; ++c)
            acc = __builtin_amdgcn_wmma_f32_16x16x32_f16(
                /*neg_a=*/false, A[c], /*neg_b=*/false, B[c],
                /*c_mod=*/(short)0, acc, /*reuse_a=*/false, /*reuse_b=*/false);

#if defined(__has_builtin)
#if __has_builtin(__builtin_amdgcn_sched_barrier)
        __builtin_amdgcn_sched_barrier(0);   // keep loop tail out of the chain
#endif
#endif

        // D layout: lanes 0-15 hold times t0..t0+7 (vgpr r = offset), lanes 16-31
        // hold t0+8..t0+15 -> one contiguous 32B store per lane.
        *(v8f*)(optr) = acc;
        bptr += WSLOTS * 16;
        optr += WSLOTS * 16;
    }
}

extern "C" void kernel_launch(void* const* d_in, const int* in_sizes, int n_in,
                              void* d_out, int out_size, void* d_ws, size_t ws_size,
                              hipStream_t stream) {
    const float* x = (const float*)d_in[0];   // (32,64,30000) fp32
    const float* h = (const float*)d_in[1];   // (413,) fp32
    float* out = (float*)d_out;               // (32,64,30000) fp32

    _Float16* hT = (_Float16*)d_ws;
    _Float16* xe = (_Float16*)((char*)d_ws + 16384);

    // 1) pack Toeplitz filter tiles (7168 halves)
    FilterBank_prep_h<<<dim3((NCHUNK * 512 + 255) / 256), dim3(256), 0, stream>>>(h, hT);

    // 2) build reflect-extended f16 signal (126 MB, L2-resident)
    FilterBank_prep_xe<<<dim3((XSTRIDE / 2 + 255) / 256, NROWS), dim3(256), 0, stream>>>(x, xe);

    // 3) WMMA FIR: 128 row tiles x (15 blocks x 8 waves) slots, each wave loops
    //    strided over the 1875 time tiles of its row tile.
    FilterBank_fir_wmma<<<dim3(NROWS / 16, 15), dim3(256), 0, stream>>>(xe, hT, out);
}
